// Seq2SeqLSTMSubspace_35175782154957
// MI455X (gfx1250) — compile-verified, matrix operand order checked by reading, not run on hardware
//
#include <hip/hip_runtime.h>
#include <hip/hip_bf16.h>

// ---------------- problem constants (match reference) ----------------
#define Bn  32
#define Sn  64
#define Tn  64
#define Vn  32000
#define En  512
#define Hn  1024
#define G4H (4 * Hn)

typedef __bf16 bf16;
typedef __attribute__((ext_vector_type(16))) __bf16 v16bf;
typedef __attribute__((ext_vector_type(8)))  __bf16 bf16x8;
typedef __attribute__((ext_vector_type(8)))  float  v8f;
typedef unsigned int v4u __attribute__((ext_vector_type(4)));
typedef int v8i __attribute__((ext_vector_type(8)));
typedef int v4i __attribute__((ext_vector_type(4)));

// Address-space-qualified int4 for the async LDS-load builtin signature:
// (global v4i*, local v4i*, imm offset, imm cpol).
typedef __attribute__((address_space(1))) v4i gv4i;
typedef __attribute__((address_space(3))) v4i lv4i;

union V16 { v16bf v; bf16x8 h[2]; };

// ---------------- CDNA5 feature probes (per-pass; host pass -> fallbacks) ---
#if __has_builtin(__builtin_amdgcn_tensor_load_to_lds)
#define HAVE_TDM 1
#endif
#if __has_builtin(__builtin_amdgcn_global_load_async_to_lds_b128)
#define HAVE_ASYNC 1
#endif

// Byte address of a __shared__ object within the workgroup LDS allocation.
__device__ __forceinline__ unsigned lds_addr_of(const void* p) {
  return (unsigned)(size_t)(__attribute__((address_space(3))) char*)(void*)p;
}

// 16B global -> LDS copy; async where the toolchain exposes it.
__device__ __forceinline__ void g2l_b128(const bf16* g, bf16* l) {
#if defined(HAVE_ASYNC)
  __builtin_amdgcn_global_load_async_to_lds_b128((gv4i*)g, (lv4i*)l, 0, 0);
#else
  *(bf16x8*)l = *(const bf16x8*)g;
#endif
}

__device__ __forceinline__ void wait_async() {
#if defined(HAVE_ASYNC)
#if __has_builtin(__builtin_amdgcn_s_wait_asynccnt)
  __builtin_amdgcn_s_wait_asynccnt(0);
#else
  asm volatile("s_wait_asynccnt 0x0" ::: "memory");
#endif
#endif
}

__device__ __forceinline__ void wait_tensor() {
#if defined(HAVE_TDM)
#if __has_builtin(__builtin_amdgcn_s_wait_tensorcnt)
  __builtin_amdgcn_s_wait_tensorcnt(0);
#else
  asm volatile("s_wait_tensorcnt 0x0" ::: "memory");
#endif
#endif
}

// TDM: contiguous 1-D copy of n8*8 bytes from global to LDS.
// D# group0: count=1, lds_addr, global_addr, type=2.
// D# group1: data_size=8B, tensor_dim0=tile_dim0=n8, dims1..=1, stride0=n8.
__device__ __forceinline__ void tdm_copy_contig(unsigned ldsAddr, const void* g,
                                                unsigned n8) {
#if defined(HAVE_TDM)
  unsigned long long ga = (unsigned long long)(size_t)g;
  v4u g0 = { 1u, ldsAddr, (unsigned)ga, (unsigned)(ga >> 32) | (2u << 30) };
  v8i g1 = { (int)0x00030000,                           // data_size=3 (8B)
             (int)((n8 & 0xFFFFu) << 16),               // tensor_dim0[15:0]
             (int)(((n8 >> 16) & 0xFFFFu) | (1u << 16)),// dim0[31:16]|tensor_dim1=1
             (int)((n8 & 0xFFFFu) << 16),               // tile_dim0
             1,                                         // tile_dim1=1
             (int)n8,                                   // tensor_dim0_stride[31:0]
             0, 0 };
  v4i z4 = { 0, 0, 0, 0 };
#if __clang_major__ >= 23
  v8i z8 = { 0, 0, 0, 0, 0, 0, 0, 0 };
  __builtin_amdgcn_tensor_load_to_lds(g0, g1, z4, z4, z8, 0);
#else
  __builtin_amdgcn_tensor_load_to_lds(g0, g1, z4, z4, 0);
#endif
#else
  (void)ldsAddr; (void)g; (void)n8;
#endif
}

// Load one WMMA 16-bit A/B operand chunk pair for this lane (ISA 7.12.2).
__device__ __forceinline__ v16bf load_tile16(const bf16* __restrict__ row,
                                             int kb, int hi) {
  V16 u;
  u.h[0] = *(const bf16x8*)(row + kb + hi * 8);
  u.h[1] = *(const bf16x8*)(row + kb + 16 + hi * 8);
  return u.v;
}

__device__ __forceinline__ v8f wmma_bf16(v16bf a, v16bf b, v8f c) {
  return __builtin_amdgcn_wmma_f32_16x16x32_bf16(false, a, false, b,
                                                 (short)0, c, false, false);
}

__device__ __forceinline__ float sigmoidf_(float x) {
  return 1.0f / (1.0f + __expf(-x));
}

// ---------------- small utility kernels ----------------
__global__ void k_f32_to_bf16(const float* __restrict__ s, bf16* __restrict__ d, int n) {
  int i = blockIdx.x * blockDim.x + threadIdx.x;
  if (i < n) d[i] = (bf16)s[i];
}

__global__ void k_zero_bf16(bf16* __restrict__ d, int n) {
  int i = blockIdx.x * blockDim.x + threadIdx.x;
  if (i < n) d[i] = (bf16)0.0f;
}

__global__ void k_zero_f32(float* __restrict__ d, int n) {
  int i = blockIdx.x * blockDim.x + threadIdx.x;
  if (i < n) d[i] = 0.0f;
}

__global__ void k_bf16_to_f32(const bf16* __restrict__ s, float* __restrict__ d, int n) {
  int i = blockIdx.x * blockDim.x + threadIdx.x;
  if (i < n) d[i] = (float)s[i];
}

__global__ void k_copy_f32(const float* __restrict__ s, float* __restrict__ d, int n) {
  int i = blockIdx.x * blockDim.x + threadIdx.x;
  if (i < n) d[i] = s[i];
}

// src_tokens [B][S] gather -> x time-major [S][B][E] bf16
__global__ void k_embed_src(const int* __restrict__ tok,
                            const float* __restrict__ emb,
                            bf16* __restrict__ out) {
  int i = blockIdx.x * blockDim.x + threadIdx.x;
  if (i >= Sn * Bn * En) return;
  int e = i % En;
  int r = i / En;
  int b = r % Bn;
  int s = r / Bn;
  int t = tok[b * Sn + s];
  out[i] = (bf16)emb[(size_t)t * En + e];
}

// relu(tgt_emb[tgt_tokens[:, :-1]]) -> [T-1][B][E] bf16 time-major
__global__ void k_embed_tgt_relu(const int* __restrict__ tok,
                                 const float* __restrict__ emb,
                                 bf16* __restrict__ out) {
  int i = blockIdx.x * blockDim.x + threadIdx.x;
  if (i >= (Tn - 1) * Bn * En) return;
  int e = i % En;
  int r = i / En;
  int b = r % Bn;
  int t = r / Bn;
  int tk = tok[b * Tn + t];
  float v = emb[(size_t)tk * En + e];
  out[i] = (bf16)(v > 0.0f ? v : 0.0f);
}

// outputs[:, 0, :] = BOS (2.0)
__global__ void k_bos_fill(float* __restrict__ out) {
  int i = blockIdx.x * blockDim.x + threadIdx.x;
  if (i >= Bn * Vn) return;
  int b = i / Vn;
  int v = i % Vn;
  out[(size_t)b * Tn * Vn + v] = 2.0f;
}

// ---------------- fused LSTM step ----------------
// G = Xt @ Wih^T + hPrev @ Whh^T (M=32, N=4096), then fused gate math.
// grid.x = H/32 WGs, 8 waves/WG; wave w -> gate w/2, half w&1.
// DIN==512 layers stage Xt into LDS (async, +16B row pad -> conflict-free
// ds_load_b128); DIN==1024 phases stream from L2-resident buffers.
template <int DIN, bool STAGE>
__global__ void __launch_bounds__(256)
k_lstm_step(const bf16* __restrict__ Xt,
            const bf16* __restrict__ hPrev,
            const bf16* __restrict__ Wih,   // [4H][DIN] bf16
            const bf16* __restrict__ Whh,   // [4H][H]   bf16
            float* __restrict__ Cst,        // [B][H] f32 state (in/out)
            bf16* __restrict__ Yt) {        // [B][H] bf16 out (= h_t)
  constexpr int LDP = DIN + 8;              // padded LDS row stride (bf16)
  __shared__ float gbuf[4][32][33];
  __shared__ bf16  sX[STAGE ? 32 * LDP : 16];

  const int tid  = threadIdx.x;
  const int wave = tid >> 5;
  const int lane = tid & 31;
  const int gate = wave >> 1;
  const int half = wave & 1;
  const int l15  = lane & 15;
  const int hi   = lane >> 4;
  const int nLocal   = half * 16 + l15;
  const int hcolBase = blockIdx.x * 32;
  const int nCol     = gate * Hn + hcolBase + half * 16;

  if constexpr (STAGE) {
    constexpr int CPR = DIN / 8;            // 16B chunks per row
    constexpr int CH  = 32 * CPR;
    for (int c = tid; c < CH; c += 256) {
      int row = c / CPR, col = c % CPR;
      g2l_b128(Xt + (size_t)row * DIN + col * 8, sX + row * LDP + col * 8);
    }
    wait_async();
    __syncthreads();
  }

  v8f acc0 = {};
  v8f acc1 = {};

  // ---- x_t @ Wih^T ----
  {
    const bf16* Brow = Wih + (size_t)(nCol + l15) * DIN;
    if constexpr (STAGE) {
      const bf16* Arow0 = sX + l15 * LDP;
      const bf16* Arow1 = sX + (16 + l15) * LDP;
      for (int kb = 0; kb < DIN; kb += 32) {
        v16bf a0 = load_tile16(Arow0, kb, hi);
        v16bf a1 = load_tile16(Arow1, kb, hi);
        v16bf b  = load_tile16(Brow,  kb, hi);
        acc0 = wmma_bf16(a0, b, acc0);
        acc1 = wmma_bf16(a1, b, acc1);
      }
    } else {
      const bf16* Arow0 = Xt + (size_t)l15 * DIN;
      const bf16* Arow1 = Xt + (size_t)(16 + l15) * DIN;
      for (int kb = 0; kb < DIN; kb += 32) {
        v16bf a0 = load_tile16(Arow0, kb, hi);
        v16bf a1 = load_tile16(Arow1, kb, hi);
        v16bf b  = load_tile16(Brow,  kb, hi);
        acc0 = wmma_bf16(a0, b, acc0);
        acc1 = wmma_bf16(a1, b, acc1);
      }
    }
  }
  // ---- h_{t-1} @ Whh^T ----
  {
    const bf16* Arow0 = hPrev + (size_t)l15 * Hn;
    const bf16* Arow1 = hPrev + (size_t)(16 + l15) * Hn;
    const bf16* Brow  = Whh + (size_t)(nCol + l15) * Hn;
    for (int kb = 0; kb < Hn; kb += 32) {
      v16bf a0 = load_tile16(Arow0, kb, hi);
      v16bf a1 = load_tile16(Arow1, kb, hi);
      v16bf b  = load_tile16(Brow,  kb, hi);
      acc0 = wmma_bf16(a0, b, acc0);
      acc1 = wmma_bf16(a1, b, acc1);
    }
  }

  // Scatter accumulators to LDS. D layout: VGPR r, lane l -> (M=r+8*hi, N=l15)
#pragma unroll
  for (int r = 0; r < 8; ++r) {
    gbuf[gate][r + 8 * hi][nLocal]      = acc0[r];
    gbuf[gate][16 + r + 8 * hi][nLocal] = acc1[r];
  }
  __syncthreads();

#pragma unroll
  for (int j = 0; j < 4; ++j) {
    int idx = tid + j * 256;
    int m   = idx >> 5;
    int nl  = idx & 31;
    float gi = gbuf[0][m][nl];
    float gf = gbuf[1][m][nl];
    float gg = gbuf[2][m][nl];
    float go = gbuf[3][m][nl];
    int hcol = hcolBase + nl;
    float c  = Cst[m * Hn + hcol];
    float cn = sigmoidf_(gf) * c + sigmoidf_(gi) * tanhf(gg);
    float hn = sigmoidf_(go) * tanhf(cn);
    Cst[m * Hn + hcol] = cn;
    Yt[m * Hn + hcol]  = (bf16)hn;
  }
}

// ---------------- output projection ----------------
// outputs[b][t+1][v] = d[t][b][:] . out_W[v][:] + out_b[v].
// The 32x1024 A-block (64KB) is shared by all 8 waves: rows 0..15 staged into
// LDS by the Tensor Data Mover, rows 16..31 by async global->LDS loads.
__global__ void __launch_bounds__(256)
k_projection(const bf16* __restrict__ Dsrc,  // [T-1][B][H] bf16
             const bf16* __restrict__ Wout,  // [V][H] bf16
             const float* __restrict__ bias, // [V]
             float* __restrict__ Out) {      // [B][T][V]
  __shared__ bf16 sA[32 * Hn];               // 64 KB
  const int tid  = threadIdx.x;
  const int wave = tid >> 5;
  const int lane = tid & 31;
  const int l15  = lane & 15;
  const int hi   = lane >> 4;
  const int nBase = blockIdx.x * 128 + wave * 16;
  const int t     = blockIdx.y;              // 0..T-2
  const int mBase = t * Bn;
  const bf16* Ag  = Dsrc + (size_t)mBase * Hn;

#if defined(HAVE_TDM)
  if (wave == 0) {
    tdm_copy_contig(lds_addr_of(sA), Ag, (16u * Hn * 2u) / 8u);  // rows 0..15
    wait_tensor();
  }
#pragma unroll
  for (int i = 0; i < 8; ++i) {              // rows 16..31: 2048 x 16B chunks
    int c = tid + i * 256;
    g2l_b128(Ag + 16 * Hn + c * 8, sA + 16 * Hn + c * 8);
  }
  wait_async();
#else
#pragma unroll
  for (int i = 0; i < 16; ++i) {             // whole block: 4096 x 16B chunks
    int c = tid + i * 256;
    g2l_b128(Ag + c * 8, sA + c * 8);
  }
  wait_async();
#endif
  __syncthreads();

  const bf16* Arow0 = sA + (size_t)l15 * Hn;
  const bf16* Arow1 = sA + (size_t)(16 + l15) * Hn;
  const bf16* Brow  = Wout + (size_t)(nBase + l15) * Hn;

  v8f acc0 = {};
  v8f acc1 = {};
  for (int kb = 0; kb < Hn; kb += 32) {
    __builtin_prefetch(Brow + kb + 256, 0, 0);   // global_prefetch_b8
    v16bf a0 = load_tile16(Arow0, kb, hi);
    v16bf a1 = load_tile16(Arow1, kb, hi);
    v16bf b  = load_tile16(Brow,  kb, hi);
    acc0 = wmma_bf16(a0, b, acc0);
    acc1 = wmma_bf16(a1, b, acc1);
  }

  const int n  = nBase + l15;
  const float bn = bias[n];
#pragma unroll
  for (int r = 0; r < 8; ++r) {
    int m0 = r + 8 * hi;        // batch rows 0..15
    int m1 = 16 + r + 8 * hi;   // batch rows 16..31
    Out[(size_t)m0 * Tn * Vn + (size_t)(t + 1) * Vn + n] = acc0[r] + bn;
    Out[(size_t)m1 * Tn * Vn + (size_t)(t + 1) * Vn + n] = acc1[r] + bn;
  }
}

// ---------------- host orchestration ----------------
static inline int cdiv(int a, int b) { return (a + b - 1) / b; }

extern "C" void kernel_launch(void* const* d_in, const int* in_sizes, int n_in,
                              void* d_out, int out_size, void* d_ws, size_t ws_size,
                              hipStream_t stream) {
  (void)in_sizes; (void)n_in; (void)out_size; (void)ws_size;

  const int*   src_tok = (const int*)d_in[0];
  const int*   tgt_tok = (const int*)d_in[1];
  const float* src_emb = (const float*)d_in[2];
  const float* tgt_emb = (const float*)d_in[3];
  const float* wf[8] = {
    (const float*)d_in[4],  (const float*)d_in[5],
    (const float*)d_in[6],  (const float*)d_in[7],
    (const float*)d_in[8],  (const float*)d_in[9],
    (const float*)d_in[10], (const float*)d_in[11]
  };
  const float* out_W = (const float*)d_in[12];
  const float* out_b = (const float*)d_in[13];
  float* out = (float*)d_out;

  char*  ws  = (char*)d_ws;
  size_t off = 0;
  auto take = [&](size_t bytes) -> char* {
    char* p = ws + off;
    off = (off + bytes + 255) & ~(size_t)255;
    return p;
  };
  bf16* xenc  = (bf16*)take((size_t)Sn * Bn * En * 2);
  bf16* xdec  = (bf16*)take((size_t)(Tn - 1) * Bn * En * 2);
  bf16* y0enc = (bf16*)take((size_t)Sn * Bn * Hn * 2);
  bf16* y1enc = (bf16*)take((size_t)Sn * Bn * Hn * 2);
  bf16* y0dec = (bf16*)take((size_t)(Tn - 1) * Bn * Hn * 2);
  bf16* y1dec = (bf16*)take((size_t)(Tn - 1) * Bn * Hn * 2);
  bf16* hzero = (bf16*)take((size_t)Bn * Hn * 2);
  float* c0   = (float*)take((size_t)Bn * Hn * 4);
  float* c1   = (float*)take((size_t)Bn * Hn * 4);

  const int wn[8] = { G4H * En, G4H * Hn, G4H * Hn, G4H * Hn,
                      G4H * En, G4H * Hn, G4H * Hn, G4H * Hn };
  bf16* wb[8];
  for (int i = 0; i < 8; ++i) wb[i] = (bf16*)take((size_t)wn[i] * 2);
  bf16* outWb = (bf16*)take((size_t)Vn * Hn * 2);

  const int TPB = 256;

  for (int i = 0; i < 8; ++i)
    k_f32_to_bf16<<<cdiv(wn[i], TPB), TPB, 0, stream>>>(wf[i], wb[i], wn[i]);
  k_f32_to_bf16<<<cdiv(Vn * Hn, TPB), TPB, 0, stream>>>(out_W, outWb, Vn * Hn);

  k_embed_src<<<cdiv(Sn * Bn * En, TPB), TPB, 0, stream>>>(src_tok, src_emb, xenc);
  k_embed_tgt_relu<<<cdiv((Tn - 1) * Bn * En, TPB), TPB, 0, stream>>>(tgt_tok, tgt_emb, xdec);

  k_zero_bf16<<<cdiv(Bn * Hn, TPB), TPB, 0, stream>>>(hzero, Bn * Hn);
  k_zero_f32 <<<cdiv(Bn * Hn, TPB), TPB, 0, stream>>>(c0, Bn * Hn);
  k_zero_f32 <<<cdiv(Bn * Hn, TPB), TPB, 0, stream>>>(c1, Bn * Hn);

  const dim3 stepGrid(Hn / 32);
  const size_t BH = (size_t)Bn * Hn;
  const size_t BE = (size_t)Bn * En;

  // encoder layer 0 (DIN=512, LDS-staged A)
  for (int t = 0; t < Sn; ++t)
    k_lstm_step<En, true><<<stepGrid, TPB, 0, stream>>>(
        xenc + t * BE, (t ? y0enc + (t - 1) * BH : hzero),
        wb[0], wb[1], c0, y0enc + t * BH);
  // encoder layer 1 (DIN=1024)
  for (int t = 0; t < Sn; ++t)
    k_lstm_step<Hn, false><<<stepGrid, TPB, 0, stream>>>(
        y0enc + t * BH, (t ? y1enc + (t - 1) * BH : hzero),
        wb[2], wb[3], c1, y1enc + t * BH);

  // decoder layer 0 (states continue from encoder finals)
  for (int t = 0; t < Tn - 1; ++t)
    k_lstm_step<En, true><<<stepGrid, TPB, 0, stream>>>(
        xdec + t * BE,
        (t ? y0dec + (t - 1) * BH : y0enc + (Sn - 1) * BH),
        wb[4], wb[5], c0, y0dec + t * BH);
  // decoder layer 1
  for (int t = 0; t < Tn - 1; ++t)
    k_lstm_step<Hn, false><<<stepGrid, TPB, 0, stream>>>(
        y0dec + t * BH,
        (t ? y1dec + (t - 1) * BH : y1enc + (Sn - 1) * BH),
        wb[6], wb[7], c1, y1dec + t * BH);

  // vocab projection + BOS row
  k_projection<<<dim3(Vn / 128, Tn - 1), TPB, 0, stream>>>(y1dec, outWb, out_b, out);
  k_bos_fill<<<cdiv(Bn * Vn, TPB), TPB, 0, stream>>>(out);

  // hidden / cell outputs
  const size_t HID = (size_t)Bn * Tn * Vn;
  const size_t CEL = HID + 2 * BH;
  k_bf16_to_f32<<<cdiv(Bn * Hn, TPB), TPB, 0, stream>>>(y0dec + (Tn - 2) * BH, out + HID, Bn * Hn);
  k_bf16_to_f32<<<cdiv(Bn * Hn, TPB), TPB, 0, stream>>>(y1dec + (Tn - 2) * BH, out + HID + BH, Bn * Hn);
  k_copy_f32  <<<cdiv(Bn * Hn, TPB), TPB, 0, stream>>>(c0, out + CEL, Bn * Hn);
  k_copy_f32  <<<cdiv(Bn * Hn, TPB), TPB, 0, stream>>>(c1, out + CEL + BH, Bn * Hn);
}
